// MambaTopDownPath_10857677324811
// MI455X (gfx1250) — matrix operand order repaired and checked by reading, not compiled, not run on hardware
//
#include <hip/hip_runtime.h>
#include <hip/hip_bf16.h>

// ---------------------------------------------------------------------------
// Problem constants
// ---------------------------------------------------------------------------
#define Bc       2
#define Cc       256
#define HT       32
#define WT       32
#define HL       64
#define WL       64
#define Lc       (HL * WL)          // 4096
#define NROW     (Bc * Lc)          // 8192
#define D_STATE  16
#define D_INNER  512
#define DT_RANK  16
#define D_CONV   4
#define DBC_N    (DT_RANK + 2 * D_STATE)   // 48

typedef __bf16 bf16;
typedef __attribute__((ext_vector_type(16))) __bf16 v16bf;
typedef __attribute__((ext_vector_type(8)))  __bf16 v8bf;
typedef __attribute__((ext_vector_type(8)))  float  v8f;

// ---------------------------------------------------------------------------
// WMMA helpers (CDNA5 16x16x32 bf16, f32 accumulate)
// ---------------------------------------------------------------------------
__device__ __forceinline__ v16bf frag_cat(v8bf lo, v8bf hi) {
  return __builtin_shufflevector(lo, hi, 0,1,2,3,4,5,6,7,8,9,10,11,12,13,14,15);
}

__device__ __forceinline__ v16bf zero_frag() {
  v16bf a;
#pragma unroll
  for (int i = 0; i < 16; ++i) a[i] = (bf16)0.0f;
  return a;
}

// A fragment: 16x32 tile of row-major bf16 A (leading dim lda elements).
// ISA 16-bit A layout: lane m = lane&15, kb = (lane&16)?8:0,
// elements 0..7 -> k = kb..kb+7, elements 8..15 -> k = kb+16..kb+23.
__device__ __forceinline__ v16bf load_a(const bf16* A, int lda, int row0, int k0) {
  int lane = threadIdx.x & 31;
  int m    = lane & 15;
  int kb   = (lane & 16) ? 8 : 0;
  const bf16* p = A + (size_t)(row0 + m) * lda + k0 + kb;
  v8bf lo = *reinterpret_cast<const v8bf*>(p);
  v8bf hi = *reinterpret_cast<const v8bf*>(p + 16);
  return frag_cat(lo, hi);
}

// B fragment from transposed weights Bt[N][ldb] (K contiguous per column n).
// ISA 16-bit B layout: lane n = lane&15, kb = (lane&16)?8:0,
// elements 0..7 -> k = kb..kb+7, elements 8..15 -> k = kb+16..kb+23.
__device__ __forceinline__ v16bf load_b(const bf16* Bt, int ldb, int col0, int k0) {
  int lane = threadIdx.x & 31;
  int n    = lane & 15;
  int kb   = (lane & 16) ? 8 : 0;
  const bf16* p = Bt + (size_t)(col0 + n) * ldb + k0 + kb;
  v8bf lo = *reinterpret_cast<const v8bf*>(p);
  v8bf hi = *reinterpret_cast<const v8bf*>(p + 16);
  return frag_cat(lo, hi);
}

__device__ __forceinline__ v8f wmma_bf16(v16bf a, v16bf b, v8f c) {
  return __builtin_amdgcn_wmma_f32_16x16x32_bf16(false, a, false, b, (short)0, c,
                                                 false, false);
}

// ---------------------------------------------------------------------------
// Prep kernels: weight transposes (f32 -> bf16, [K][N] -> [N][Kp]), misc
// ---------------------------------------------------------------------------
__global__ void __launch_bounds__(256, 1)
k_wT(const float* __restrict__ W, bf16* __restrict__ Wt, int K, int N, int Kp) {
  size_t i = (size_t)blockIdx.x * 256 + threadIdx.x;
  if (i >= (size_t)N * Kp) return;
  int k = (int)(i % Kp);
  int n = (int)(i / Kp);
  Wt[i] = (k < K) ? (bf16)W[(size_t)k * N + n] : (bf16)0.0f;
}

// fuse_w [O=256][I=512][3][3] -> Wc[tap][co][ci] bf16
__global__ void __launch_bounds__(256, 1)
k_convwT(const float* __restrict__ Fw, bf16* __restrict__ Wc) {
  size_t i = (size_t)blockIdx.x * 256 + threadIdx.x;
  if (i >= (size_t)9 * 256 * 512) return;
  int ci = (int)(i & 511);
  int co = (int)((i >> 9) & 255);
  int t  = (int)(i >> 17);
  Wc[i] = (bf16)Fw[((size_t)co * 512 + ci) * 9 + t];
}

// A = -exp(A_log); folded BN scale/bias
__global__ void __launch_bounds__(256, 1)
k_misc(const float* __restrict__ A_log,
       const float* __restrict__ gamma, const float* __restrict__ beta,
       const float* __restrict__ mean, const float* __restrict__ var,
       const float* __restrict__ fuse_b,
       float* __restrict__ Aneg, float* __restrict__ scale,
       float* __restrict__ bias) {
  int i = blockIdx.x * 256 + threadIdx.x;
  if (i < D_INNER * D_STATE) Aneg[i] = -__expf(A_log[i]);
  if (i < Cc) {
    float sc = gamma[i] * rsqrtf(var[i] + 1e-5f);
    scale[i] = sc;
    bias[i]  = beta[i] + (fuse_b[i] - mean[i]) * sc;
  }
}

// u[r][c] = top (nearest-upsampled), bf16 row-major [8192][256]
__global__ void __launch_bounds__(256, 1)
k_build_u(const float* __restrict__ top, bf16* __restrict__ U) {
  size_t i = (size_t)blockIdx.x * 256 + threadIdx.x;
  if (i >= (size_t)NROW * Cc) return;
  int c = (int)(i & 255);
  size_t r = i >> 8;
  int l = (int)(r & (Lc - 1));
  int b = (int)(r >> 12);
  int hp = l >> 6, wp = l & 63;
  U[i] = (bf16)top[(((size_t)b * Cc + c) * HT + (hp >> 1)) * WT + (wp >> 1)];
}

// lateral NCHW f32 -> NHWC bf16 into cat channels [256,512)
__global__ void __launch_bounds__(256, 1)
k_lateral(const float* __restrict__ lat, bf16* __restrict__ Cat) {
  size_t i = (size_t)blockIdx.x * 256 + threadIdx.x;
  if (i >= (size_t)Bc * Lc * Cc) return;
  int c = (int)(i & 255);
  size_t p = i >> 8;                 // b*4096 + h*64 + w
  int b = (int)(p >> 12);
  int hw = (int)(p & (Lc - 1));
  Cat[p * 512 + 256 + c] = (bf16)lat[((size_t)b * Cc + c) * Lc + hw];
}

// ---------------------------------------------------------------------------
// GEMM 1: xz = u @ W_in   (M=8192, K=256, N=1024) -> X, Z f32
// block = 128 thr (4 waves along M); wave tile 16x64
// ---------------------------------------------------------------------------
__global__ void __launch_bounds__(128, 1)
k_in_proj(const bf16* __restrict__ U, const bf16* __restrict__ WinT,
          float* __restrict__ X, float* __restrict__ Z) {
  int wave = threadIdx.x >> 5;
  int row0 = blockIdx.x * 64 + wave * 16;
  int col0 = blockIdx.y * 64;
  v8f acc[4];
#pragma unroll
  for (int j = 0; j < 4; ++j)
#pragma unroll
    for (int r = 0; r < 8; ++r) acc[j][r] = 0.0f;
  for (int k = 0; k < Cc; k += 32) {
    v16bf a = load_a(U, Cc, row0, k);
#pragma unroll
    for (int j = 0; j < 4; ++j) {
      v16bf bfr = load_b(WinT, Cc, col0 + 16 * j, k);
      acc[j] = wmma_bf16(a, bfr, acc[j]);
    }
  }
  int lane = threadIdx.x & 31;
  int n  = lane & 15;
  int mb = (lane & 16) ? 8 : 0;
#pragma unroll
  for (int j = 0; j < 4; ++j)
#pragma unroll
    for (int r = 0; r < 8; ++r) {
      int row = row0 + mb + r;
      int col = col0 + 16 * j + n;
      float v = acc[j][r];
      if (col < D_INNER) X[(size_t)row * D_INNER + col] = v;
      else               Z[(size_t)row * D_INNER + col - D_INNER] = v;
    }
}

// ---------------------------------------------------------------------------
// Depthwise causal conv1d (k=4) + SiLU.  X [8192][512] -> XC f32 + XCh bf16
// ---------------------------------------------------------------------------
__global__ void __launch_bounds__(256, 1)
k_dwconv(const float* __restrict__ X, const float* __restrict__ cw,
         const float* __restrict__ cb,
         float* __restrict__ XC, bf16* __restrict__ XCh) {
  size_t i = (size_t)blockIdx.x * 256 + threadIdx.x;
  if (i >= (size_t)NROW * D_INNER) return;
  int d = (int)(i & (D_INNER - 1));
  size_t r = i >> 9;
  int l = (int)(r & (Lc - 1));
  float acc = cb[d];
#pragma unroll
  for (int k = 0; k < D_CONV; ++k) {
    int ls = l + k - (D_CONV - 1);
    if (ls >= 0) acc += cw[d * D_CONV + k] * X[(r + k - (D_CONV - 1)) * D_INNER + d];
  }
  float s = acc / (1.0f + __expf(-acc));
  XC[i]  = s;
  XCh[i] = (bf16)s;
}

// ---------------------------------------------------------------------------
// GEMM 2: dbc = x_conv @ W_x  (M=8192, K=512, N=48) -> DBC f32
// ---------------------------------------------------------------------------
__global__ void __launch_bounds__(128, 1)
k_xproj(const bf16* __restrict__ XCh, const bf16* __restrict__ WxT,
        float* __restrict__ DBC) {
  int wave = threadIdx.x >> 5;
  int row0 = blockIdx.x * 64 + wave * 16;
  v8f acc[3];
#pragma unroll
  for (int j = 0; j < 3; ++j)
#pragma unroll
    for (int r = 0; r < 8; ++r) acc[j][r] = 0.0f;
  for (int k = 0; k < D_INNER; k += 32) {
    v16bf a = load_a(XCh, D_INNER, row0, k);
#pragma unroll
    for (int j = 0; j < 3; ++j) {
      v16bf bfr = load_b(WxT, D_INNER, 16 * j, k);
      acc[j] = wmma_bf16(a, bfr, acc[j]);
    }
  }
  int lane = threadIdx.x & 31;
  int n  = lane & 15;
  int mb = (lane & 16) ? 8 : 0;
#pragma unroll
  for (int j = 0; j < 3; ++j)
#pragma unroll
    for (int r = 0; r < 8; ++r)
      DBC[(size_t)(row0 + mb + r) * DBC_N + 16 * j + n] = acc[j][r];
}

// ---------------------------------------------------------------------------
// GEMM 3: dt = softplus(dbc[:, :16] @ W_dt + b_dt)  (M=8192, K=16->32, N=512)
// A loaded from f32 DBC with on-the-fly bf16 convert; K padded with zeros
// (elements 8..15 of the fragment map to k>=16 for both half-waves).
// ---------------------------------------------------------------------------
__global__ void __launch_bounds__(128, 1)
k_dtproj(const float* __restrict__ DBC, const bf16* __restrict__ WdtT,
         const float* __restrict__ b_dt, float* __restrict__ DT) {
  int wave = threadIdx.x >> 5;
  int row0 = blockIdx.x * 64 + wave * 16;
  int col0 = blockIdx.y * 64;
  int lane = threadIdx.x & 31;
  int m    = lane & 15;
  int kb   = (lane & 16) ? 8 : 0;
  const float* p = DBC + (size_t)(row0 + m) * DBC_N + kb;
  v16bf a = zero_frag();
#pragma unroll
  for (int i = 0; i < 8; ++i) a[i] = (bf16)p[i];   // k = kb..kb+7 (< 16)
  v8f acc[4];
#pragma unroll
  for (int j = 0; j < 4; ++j) {
#pragma unroll
    for (int r = 0; r < 8; ++r) acc[j][r] = 0.0f;
    v16bf bfr = load_b(WdtT, 32, col0 + 16 * j, 0);
    acc[j] = wmma_bf16(a, bfr, acc[j]);
  }
  int n  = lane & 15;
  int mb = (lane & 16) ? 8 : 0;
#pragma unroll
  for (int j = 0; j < 4; ++j)
#pragma unroll
    for (int r = 0; r < 8; ++r) {
      int col = col0 + 16 * j + n;
      float v = acc[j][r] + b_dt[col];
      float sp = (v > 20.0f) ? v : __logf(1.0f + __expf(v));
      DT[(size_t)(row0 + mb + r) * D_INNER + col] = sp;
    }
}

// ---------------------------------------------------------------------------
// Selective scan. One lane per (d_local, s): wave = 2 channels x 16 states.
// Block = 256 thr = 16 channels; grid = B * (512/16) = 64 blocks.
// h_{l} = exp(dt*A)*h_{l-1} + dt*x*Bm ;  y = sum_s h*Cm ; y=(y+D*x)*silu(z)
// ---------------------------------------------------------------------------
__global__ void __launch_bounds__(256, 1)
k_scan(const float* __restrict__ DT, const float* __restrict__ XC,
       const float* __restrict__ DBC, const float* __restrict__ Zb,
       const float* __restrict__ Aneg, const float* __restrict__ Dv,
       bf16* __restrict__ Y) {
  __shared__ float sdt[64][16];
  __shared__ float sx[64][16];
  __shared__ float sB[64][16];
  __shared__ float sC[64][16];
  int b  = blockIdx.x >> 5;
  int d0 = (blockIdx.x & 31) * 16;
  int tid  = threadIdx.x;
  int wave = tid >> 5;
  int lane = tid & 31;
  int dl = wave * 2 + (lane >> 4);     // 0..15 local channel
  int d  = d0 + dl;
  int s  = lane & 15;
  float a_ds = Aneg[d * D_STATE + s];
  float Dd   = Dv[d];
  float h = 0.0f;
  for (int l0 = 0; l0 < Lc; l0 += 64) {
    for (int t = tid; t < 64 * 16; t += 256) {
      int i = t >> 4, c = t & 15;
      size_t r = (size_t)b * Lc + l0 + i;
      sdt[i][c] = DT[r * D_INNER + d0 + c];
      sx[i][c]  = XC[r * D_INNER + d0 + c];
      sB[i][c]  = DBC[r * DBC_N + DT_RANK + c];
      sC[i][c]  = DBC[r * DBC_N + DT_RANK + D_STATE + c];
    }
    __syncthreads();
    for (int i = 0; i < 64; ++i) {
      float dtv = sdt[i][dl];
      float xv  = sx[i][dl];
      h = __expf(dtv * a_ds) * h + dtv * xv * sB[i][s];
      float part = h * sC[i][s];
      part += __shfl_xor(part, 1, 32);
      part += __shfl_xor(part, 2, 32);
      part += __shfl_xor(part, 4, 32);
      part += __shfl_xor(part, 8, 32);
      if (s == 0) {
        size_t r = (size_t)b * Lc + l0 + i;
        float yv = part + Dd * xv;
        float zv = Zb[r * D_INNER + d];
        yv *= zv / (1.0f + __expf(-zv));
        Y[r * D_INNER + d] = (bf16)yv;
      }
    }
    __syncthreads();
  }
}

// ---------------------------------------------------------------------------
// GEMM 4: m = y @ W_out (M=8192, K=512, N=256); epilogue: + top_up, write
// NHWC bf16 cat channels [0,256).
// ---------------------------------------------------------------------------
__global__ void __launch_bounds__(128, 1)
k_out_proj(const bf16* __restrict__ Y, const bf16* __restrict__ WoutT,
           const float* __restrict__ top, bf16* __restrict__ Cat) {
  int wave = threadIdx.x >> 5;
  int row0 = blockIdx.x * 64 + wave * 16;
  int col0 = blockIdx.y * 64;
  v8f acc[4];
#pragma unroll
  for (int j = 0; j < 4; ++j)
#pragma unroll
    for (int r = 0; r < 8; ++r) acc[j][r] = 0.0f;
  for (int k = 0; k < D_INNER; k += 32) {
    v16bf a = load_a(Y, D_INNER, row0, k);
#pragma unroll
    for (int j = 0; j < 4; ++j) {
      v16bf bfr = load_b(WoutT, D_INNER, col0 + 16 * j, k);
      acc[j] = wmma_bf16(a, bfr, acc[j]);
    }
  }
  int lane = threadIdx.x & 31;
  int n  = lane & 15;
  int mb = (lane & 16) ? 8 : 0;
#pragma unroll
  for (int j = 0; j < 4; ++j)
#pragma unroll
    for (int r = 0; r < 8; ++r) {
      int row = row0 + mb + r;
      int col = col0 + 16 * j + n;
      int b = row >> 12;
      int l = row & (Lc - 1);
      int hp = l >> 6, wp = l & 63;
      float tu = top[(((size_t)b * Cc + col) * HT + (hp >> 1)) * WT + (wp >> 1)];
      Cat[((size_t)b * Lc + l) * 512 + col] = (bf16)(acc[j][r] + tu);
    }
}

// ---------------------------------------------------------------------------
// Fused 3x3 conv (512->256) + BN + ReLU via implicit GEMM.
// Block = 128 thr (4 waves along N); one M-tile = 16 pixels along w of one row.
// K loop: 9 taps x (512/32) ksteps. grid = B*H*(W/16) = 512 blocks.
// ---------------------------------------------------------------------------
__global__ void __launch_bounds__(128, 1)
k_fuse_conv(const bf16* __restrict__ Cat, const bf16* __restrict__ Wc,
            const float* __restrict__ scale, const float* __restrict__ bias,
            float* __restrict__ Out) {
  int mt = blockIdx.x;
  int w0 = (mt & 3) * 16;
  int h  = (mt >> 2) & 63;
  int b  = mt >> 8;
  int wave = threadIdx.x >> 5;
  int col0 = wave * 64;
  int lane = threadIdx.x & 31;
  int m  = lane & 15;
  int kb = (lane & 16) ? 8 : 0;
  v8f acc[4];
#pragma unroll
  for (int j = 0; j < 4; ++j)
#pragma unroll
    for (int r = 0; r < 8; ++r) acc[j][r] = 0.0f;
#pragma unroll
  for (int t = 0; t < 9; ++t) {
    int kh = t / 3, kw = t % 3;
    int hi = h + kh - 1;
    int wi = w0 + m + kw - 1;
    bool ok = (hi >= 0) & (hi < HL) & (wi >= 0) & (wi < WL);
    const bf16* arow = Cat + (((size_t)b * HL + hi) * WL + wi) * 512;
    const bf16* wt   = Wc + (size_t)t * 256 * 512;
    for (int k = 0; k < 512; k += 32) {
      v16bf a;
      if (ok) {
        const bf16* p = arow + k + kb;
        v8bf lo = *reinterpret_cast<const v8bf*>(p);
        v8bf hi2 = *reinterpret_cast<const v8bf*>(p + 16);
        a = frag_cat(lo, hi2);
        __builtin_prefetch(p + 32, 0, 0);
      } else {
        a = zero_frag();
      }
#pragma unroll
      for (int j = 0; j < 4; ++j) {
        v16bf bfr = load_b(wt, 512, col0 + 16 * j, k);
        acc[j] = wmma_bf16(a, bfr, acc[j]);
      }
    }
  }
  int n  = lane & 15;
  int mb = (lane & 16) ? 8 : 0;
#pragma unroll
  for (int j = 0; j < 4; ++j)
#pragma unroll
    for (int r = 0; r < 8; ++r) {
      int co = col0 + 16 * j + n;
      int w  = w0 + mb + r;
      float v = acc[j][r] * scale[co] + bias[co];
      Out[(((size_t)b * Cc + co) * HL + h) * WL + w] = v > 0.0f ? v : 0.0f;
    }
}

// ---------------------------------------------------------------------------
// Launch
// ---------------------------------------------------------------------------
extern "C" void kernel_launch(void* const* d_in, const int* in_sizes, int n_in,
                              void* d_out, int out_size, void* d_ws, size_t ws_size,
                              hipStream_t stream) {
  (void)in_sizes; (void)n_in; (void)out_size; (void)ws_size;
  const float* top     = (const float*)d_in[0];
  const float* lateral = (const float*)d_in[1];
  const float* W_in    = (const float*)d_in[2];
  const float* conv_w  = (const float*)d_in[3];
  const float* conv_b  = (const float*)d_in[4];
  const float* W_x     = (const float*)d_in[5];
  const float* W_dt    = (const float*)d_in[6];
  const float* b_dt    = (const float*)d_in[7];
  const float* A_log   = (const float*)d_in[8];
  const float* Dvec    = (const float*)d_in[9];
  const float* W_out   = (const float*)d_in[10];
  const float* fuse_w  = (const float*)d_in[11];
  const float* fuse_b  = (const float*)d_in[12];
  const float* bn_g    = (const float*)d_in[13];
  const float* bn_b    = (const float*)d_in[14];
  const float* bn_m    = (const float*)d_in[15];
  const float* bn_v    = (const float*)d_in[16];
  float* out = (float*)d_out;

  char* ws = (char*)d_ws;
  size_t off = 0;
  auto alloc = [&](size_t bytes) {
    size_t o = off;
    off = (off + bytes + 255) & ~(size_t)255;
    return o;
  };
  bf16*  U     = (bf16*)(ws + alloc((size_t)NROW * Cc * 2));
  bf16*  WinT  = (bf16*)(ws + alloc((size_t)1024 * 256 * 2));
  bf16*  WxT   = (bf16*)(ws + alloc((size_t)48 * 512 * 2));
  bf16*  WdtT  = (bf16*)(ws + alloc((size_t)512 * 32 * 2));
  bf16*  WoutT = (bf16*)(ws + alloc((size_t)256 * 512 * 2));
  bf16*  WcT   = (bf16*)(ws + alloc((size_t)9 * 256 * 512 * 2));
  float* Aneg  = (float*)(ws + alloc((size_t)D_INNER * D_STATE * 4));
  float* bnS   = (float*)(ws + alloc((size_t)Cc * 4));
  float* bnB   = (float*)(ws + alloc((size_t)Cc * 4));
  float* X     = (float*)(ws + alloc((size_t)NROW * D_INNER * 4));
  float* Z     = (float*)(ws + alloc((size_t)NROW * D_INNER * 4));
  float* XC    = (float*)(ws + alloc((size_t)NROW * D_INNER * 4));
  bf16*  XCh   = (bf16*)(ws + alloc((size_t)NROW * D_INNER * 2));
  float* DBC   = (float*)(ws + alloc((size_t)NROW * DBC_N * 4));
  float* DT    = (float*)(ws + alloc((size_t)NROW * D_INNER * 4));
  bf16*  Y     = (bf16*)(ws + alloc((size_t)NROW * D_INNER * 2));
  bf16*  Cat   = (bf16*)(ws + alloc((size_t)NROW * 512 * 2));

  // --- prep ---
  k_wT<<<dim3((1024 * 256 + 255) / 256), dim3(256), 0, stream>>>(W_in, WinT, 256, 1024, 256);
  k_wT<<<dim3((48 * 512 + 255) / 256), dim3(256), 0, stream>>>(W_x, WxT, 512, 48, 512);
  k_wT<<<dim3((512 * 32 + 255) / 256), dim3(256), 0, stream>>>(W_dt, WdtT, 16, 512, 32);
  k_wT<<<dim3((256 * 512 + 255) / 256), dim3(256), 0, stream>>>(W_out, WoutT, 512, 256, 512);
  k_convwT<<<dim3((9 * 256 * 512 + 255) / 256), dim3(256), 0, stream>>>(fuse_w, WcT);
  k_misc<<<dim3(32), dim3(256), 0, stream>>>(A_log, bn_g, bn_b, bn_m, bn_v, fuse_b,
                                             Aneg, bnS, bnB);
  k_build_u<<<dim3((NROW * Cc + 255) / 256), dim3(256), 0, stream>>>(top, U);
  k_lateral<<<dim3((Bc * Lc * Cc + 255) / 256), dim3(256), 0, stream>>>(lateral, Cat);

  // --- mamba path ---
  k_in_proj<<<dim3(NROW / 64, 1024 / 64), dim3(128), 0, stream>>>(U, WinT, X, Z);
  k_dwconv<<<dim3((NROW * D_INNER + 255) / 256), dim3(256), 0, stream>>>(X, conv_w, conv_b,
                                                                         XC, XCh);
  k_xproj<<<dim3(NROW / 64), dim3(128), 0, stream>>>(XCh, WxT, DBC);
  k_dtproj<<<dim3(NROW / 64, D_INNER / 64), dim3(128), 0, stream>>>(DBC, WdtT, b_dt, DT);
  k_scan<<<dim3(Bc * (D_INNER / 16)), dim3(256), 0, stream>>>(DT, XC, DBC, Z, Aneg, Dvec, Y);
  k_out_proj<<<dim3(NROW / 64, Cc / 64), dim3(128), 0, stream>>>(Y, WoutT, top, Cat);

  // --- fuse conv + BN + ReLU ---
  k_fuse_conv<<<dim3(Bc * HL * (WL / 16)), dim3(128), 0, stream>>>(Cat, WcT, bnS, bnB, out);
}
